// Block_89739046682999
// MI455X (gfx1250) — compile-verified, matrix operand order checked by reading, not run on hardware
//
#include <hip/hip_runtime.h>

// ---------------------------------------------------------------------------
// Types for CDNA5 WMMA (wave32) + TDM descriptors.
// ---------------------------------------------------------------------------
typedef __bf16 bf16;
typedef __bf16 v8bf  __attribute__((ext_vector_type(8)));
typedef __bf16 v16bf __attribute__((ext_vector_type(16)));
typedef float  v8f   __attribute__((ext_vector_type(8)));
typedef unsigned int u32x4 __attribute__((ext_vector_type(4)));
typedef int          i32x4 __attribute__((ext_vector_type(4)));
typedef int          i32x8 __attribute__((ext_vector_type(8)));

union Acc { v8f v; float f[8]; };

#define WMMA_B16(A_, B_, C_) \
  __builtin_amdgcn_wmma_f32_16x16x32_bf16(false, (A_), false, (B_), (short)0, (C_), false, false)

// Intra-wave LDS ordering (writes by some lanes, reads by others, same wave).
#define LDS_FENCE() asm volatile("s_wait_dscnt 0" ::: "memory")

// Build a 16-wide bf16 fragment from two contiguous 16-byte chunks.
// ISA A/B fragment layout per lane: K = khalf+{0..7}, then khalf+16+{0..7}.
static __device__ inline v16bf ld16(const bf16* p0, const bf16* p1) {
  union { v16bf v; v8bf h[2]; } u;
  u.h[0] = *(const v8bf*)p0;
  u.h[1] = *(const v8bf*)p1;
  return u.v;
}

// ---------------------------------------------------------------------------
// Tensor Data Mover: async 2D bf16 tile load global -> LDS (ISA 08, D# enc).
// tile_d0 = elements per row, tile_d1 = rows, stride_el = row stride (elems).
// Issue from one wave; completion via s_wait_tensorcnt + workgroup barrier.
// ---------------------------------------------------------------------------
static __device__ inline void tdm_load_2d_bf16(unsigned lds_off, const bf16* gaddr,
                                               unsigned tile_d0, unsigned tile_d1,
                                               unsigned stride_el) {
  unsigned long long ga = (unsigned long long)(size_t)gaddr;
  unsigned td0 = stride_el;      // tensor_dim0 (never OOB for our exact tiles)
  unsigned td1 = 0x000FFFFFu;    // tensor_dim1: large (no OOB)
  u32x4 g0;
  g0[0] = 1u;                                            // count=1, user mode
  g0[1] = lds_off;                                       // lds_addr (bytes)
  g0[2] = (unsigned)ga;                                  // global_addr[31:0]
  g0[3] = (unsigned)((ga >> 32) & 0x01ffffffull) | 0x80000000u;  // addr hi | type=2
  i32x8 g1;
  g1[0] = 1 << 16;                                       // data_size = 2 bytes
  g1[1] = (int)((td0 & 0xffffu) << 16);                  // tensor_dim0[15:0]
  g1[2] = (int)((td0 >> 16) | ((td1 & 0xffffu) << 16));  // dim0 hi | dim1 lo
  g1[3] = (int)((td1 >> 16) | (tile_d0 << 16));          // dim1 hi | tile_dim0
  g1[4] = (int)(tile_d1 & 0xffffu);                      // tile_dim1 (tile_dim2=0)
  g1[5] = (int)stride_el;                                // tensor_dim0_stride lo
  g1[6] = 0;                                             // stride hi / dim1_stride
  g1[7] = 0;
  i32x4 z4 = {0, 0, 0, 0};
  i32x8 z8 = {0, 0, 0, 0, 0, 0, 0, 0};
  // amdgpu-toolchain (clang-23) 6-arg form.
  __builtin_amdgcn_tensor_load_to_lds(g0, g1, z4, z4, z8, 0);
}

// ---------------------------------------------------------------------------
// fp32 -> bf16 conversion (plain weights)
// ---------------------------------------------------------------------------
__global__ void f2bf_kernel(const float* __restrict__ in, bf16* __restrict__ out, size_t n) {
  size_t i = (size_t)blockIdx.x * 256 + threadIdx.x;
  size_t stride = (size_t)gridDim.x * 256;
  for (; i < n; i += stride) out[i] = (bf16)in[i];
}

// Wq/Wk/Wv: [H][C][D] fp32 -> [C][H*D] bf16 (so QKV is one plain GEMM).
__global__ void wqkv_cvt_kernel(const float* __restrict__ in, bf16* __restrict__ out,
                                int C, int D, int HD) {
  size_t n = (size_t)C * HD;
  size_t i = (size_t)blockIdx.x * 256 + threadIdx.x;
  size_t stride = (size_t)gridDim.x * 256;
  for (; i < n; i += stride) {
    int h = (int)(i / ((size_t)C * D));
    size_t r = i - (size_t)h * C * D;
    int c = (int)(r / D);
    int d = (int)(r - (size_t)c * D);
    out[(size_t)c * HD + h * D + d] = (bf16)in[i];
  }
}

// ---------------------------------------------------------------------------
// LayerNorm: one block per row, 256 threads, bf16 output.
// ---------------------------------------------------------------------------
__global__ void ln_bf16_kernel(const float* __restrict__ x, const float* __restrict__ g,
                               const float* __restrict__ be, bf16* __restrict__ out, int C) {
  __shared__ float s1[256];
  __shared__ float s2[256];
  int row = blockIdx.x, tid = threadIdx.x;
  const float* xr = x + (size_t)row * C;
  float s = 0.f, ss = 0.f;
  for (int c = tid; c < C; c += 256) { float v = xr[c]; s += v; ss += v * v; }
  s1[tid] = s; s2[tid] = ss;
  __syncthreads();
  for (int off = 128; off > 0; off >>= 1) {
    if (tid < off) { s1[tid] += s1[tid + off]; s2[tid] += s2[tid + off]; }
    __syncthreads();
  }
  float mean = s1[0] / C;
  float var  = s2[0] / C - mean * mean;
  float rs   = rsqrtf(var + 1e-5f);
  bf16* o = out + (size_t)row * C;
  for (int c = tid; c < C; c += 256)
    o[c] = (bf16)((xr[c] - mean) * rs * g[c] + be[c]);
}

// ---------------------------------------------------------------------------
// bf16 WMMA GEMM: C[M,N] = A[M,K]*B[K,N] (+bias)(+relu)(+resid)
// 256 threads = 8 waves; 128x128 macro-tile; BK = 32 (one WMMA K-step).
// Wave tile 32x64 -> 8 WMMA per K-step from 6 fragments.
// A tile: TDM async DMA (double-buffered). B tile: transpose-on-store to LDS.
// ---------------------------------------------------------------------------
__global__ void gemm_wmma_bf16(const bf16* __restrict__ A, const bf16* __restrict__ Bw,
                               const float* __restrict__ bias, const float* __restrict__ resid,
                               float* __restrict__ outF, bf16* __restrict__ outB,
                               int M, int N, int K, int relu) {
  __shared__ __align__(16) bf16 As[2][128 * 32];  // [row][k]
  __shared__ __align__(16) bf16 Bt[2][128 * 32];  // transposed: [n][k]
  int tid = threadIdx.x, lane = tid & 31, w = tid >> 5;
  int half = lane >> 4, l15 = lane & 15, khalf = half * 8;
  int m0 = blockIdx.y * 128, n0 = blockIdx.x * 128;
  int wr = (w & 3) * 32;     // wave row tile within 128
  int wc = (w >> 2) * 64;    // wave col group within 128

  v8f zero = {0.f, 0.f, 0.f, 0.f, 0.f, 0.f, 0.f, 0.f};
  Acc acc[2][4];
#pragma unroll
  for (int r = 0; r < 2; ++r)
#pragma unroll
    for (int c = 0; c < 4; ++c) acc[r][c].v = zero;

  auto stageB = [&](int k0, bf16* dst) {
    int kr = tid >> 3, c16 = (tid & 7) * 16;
    const bf16* src = &Bw[(size_t)(k0 + kr) * N + n0 + c16];
    v8bf b0 = *(const v8bf*)src;
    v8bf b1 = *(const v8bf*)(src + 8);
#pragma unroll
    for (int j = 0; j < 8; ++j) dst[(c16 + j) * 32 + kr] = b0[j];
#pragma unroll
    for (int j = 0; j < 8; ++j) dst[(c16 + 8 + j) * 32 + kr] = b1[j];
  };

  // prologue: kick off first A tile DMA + stage first B tile
  if (w == 0)
    tdm_load_2d_bf16((unsigned)(size_t)&As[0][0], &A[(size_t)m0 * K], 32, 128, (unsigned)K);
  stageB(0, &Bt[0][0]);

  for (int k0 = 0; k0 < K; k0 += 32) {
    int p = (k0 >> 5) & 1;
    if (w == 0) __builtin_amdgcn_s_wait_tensorcnt(0);  // A[p] DMA complete
    __syncthreads();
    if (k0 + 32 < K) {  // prefetch next tiles into the other buffer
      if (w == 0)
        tdm_load_2d_bf16((unsigned)(size_t)&As[p ^ 1][0],
                         &A[(size_t)m0 * K + (k0 + 32)], 32, 128, (unsigned)K);
      stageB(k0 + 32, &Bt[p ^ 1][0]);
    }
    const bf16* Ab = &As[p][0];
    const bf16* Bb = &Bt[p][0];
    const bf16* a0p = &Ab[(wr + l15) * 32];
    const bf16* a1p = &Ab[(wr + 16 + l15) * 32];
    v16bf ar0 = ld16(a0p + khalf, a0p + khalf + 16);
    v16bf ar1 = ld16(a1p + khalf, a1p + khalf + 16);
#pragma unroll
    for (int nt = 0; nt < 4; ++nt) {
      const bf16* bp = &Bb[(wc + nt * 16 + l15) * 32];
      v16bf bfv = ld16(bp + khalf, bp + khalf + 16);
      acc[0][nt].v = WMMA_B16(ar0, bfv, acc[0][nt].v);
      acc[1][nt].v = WMMA_B16(ar1, bfv, acc[1][nt].v);
    }
  }

#pragma unroll
  for (int rt = 0; rt < 2; ++rt)
#pragma unroll
    for (int nt = 0; nt < 4; ++nt) {
      int gn = n0 + wc + nt * 16 + l15;
#pragma unroll
      for (int g = 0; g < 8; ++g) {
        int gm = m0 + wr + rt * 16 + g + 8 * half;
        float v = acc[rt][nt].f[g];
        if (bias)  v += bias[gn];
        if (relu)  v = fmaxf(v, 0.f);
        if (resid) v += resid[(size_t)gm * N + gn];
        if (outF)  outF[(size_t)gm * N + gn] = v;
        if (outB)  outB[(size_t)gm * N + gn] = (bf16)v;
      }
    }
}

// ---------------------------------------------------------------------------
// Causal flash attention, bf16 WMMA, D = 64, Q/K/V in [B,T,C] with head offset.
// grid.x = B*H, grid.y = T/128; 8 waves, each owns a 16-row Q tile.
// K tile (B^T layout as stored) via TDM double-buffer; V transposed on store.
// ---------------------------------------------------------------------------
__global__ void attn_flash_kernel(const bf16* __restrict__ Q, const bf16* __restrict__ Kg,
                                  const bf16* __restrict__ Vg, bf16* __restrict__ outA,
                                  int T, int C, int Hh) {
  __shared__ __align__(16) bf16  Ks[2][32 * 64];   // [s][d]  (== B^T layout for QK^T)
  __shared__ __align__(16) bf16  Vt[2][64 * 32];   // [d][s]  (== B^T layout for PV)
  __shared__ __align__(16) float Ss[8][16 * 32];   // per-wave score tile
  __shared__ __align__(16) bf16  Ps[8][16 * 32];   // per-wave exp(P) tile
  __shared__ float alph[8][16];

  int tid = threadIdx.x, lane = tid & 31, w = tid >> 5;
  int half = lane >> 4, l15 = lane & 15, khalf = half * 8;
  int bh = blockIdx.x;
  int b = bh / Hh, h = bh - b * Hh;
  int qb = blockIdx.y * 128;
  int q0 = qb + w * 16;

  const bf16* Qh = Q  + ((size_t)b * T) * C + h * 64;
  const bf16* Kh = Kg + ((size_t)b * T) * C + h * 64;
  const bf16* Vh = Vg + ((size_t)b * T) * C + h * 64;

  // Q fragments, resident for the whole kernel (K-dim d = 0..63 -> 2 frags).
  const bf16* qrow = Qh + (size_t)(q0 + l15) * C;
  v16bf aq0 = ld16(qrow + khalf,      qrow + khalf + 16);
  v16bf aq1 = ld16(qrow + 32 + khalf, qrow + 48 + khalf);

  v8f zero = {0.f, 0.f, 0.f, 0.f, 0.f, 0.f, 0.f, 0.f};
  Acc o[4];
#pragma unroll
  for (int t = 0; t < 4; ++t) o[t].v = zero;
  float mrow = -3.0e38f, lsum = 0.f;

  auto stageV = [&](int kt0, bf16* dst) {
    int s = tid >> 3, c8 = (tid & 7) * 8;
    v8bf vv = *(const v8bf*)&Vh[(size_t)(kt0 + s) * C + c8];
#pragma unroll
    for (int j = 0; j < 8; ++j) dst[(c8 + j) * 32 + s] = vv[j];
  };

  int ktend = qb + 128;
  if (w == 0)
    tdm_load_2d_bf16((unsigned)(size_t)&Ks[0][0], Kh, 64, 32, (unsigned)C);
  stageV(0, &Vt[0][0]);

  for (int kt0 = 0; kt0 < ktend; kt0 += 32) {
    int p = (kt0 >> 5) & 1;
    if (w == 0) __builtin_amdgcn_s_wait_tensorcnt(0);
    __syncthreads();
    if (kt0 + 32 < ktend) {
      if (w == 0)
        tdm_load_2d_bf16((unsigned)(size_t)&Ks[p ^ 1][0],
                         Kh + (size_t)(kt0 + 32) * C, 64, 32, (unsigned)C);
      stageV(kt0 + 32, &Vt[p ^ 1][0]);
    }
    if (kt0 > q0 + 15) continue;  // wave-uniform: fully masked for this Q tile

    // ---- S = Q K^T for 32 keys (two 16-wide N tiles, K-dim 64 = 2 steps)
    Acc s0, s1; s0.v = zero; s1.v = zero;
    {
      const bf16* kr0 = &Ks[p][(size_t)l15 * 64];
      const bf16* kr1 = &Ks[p][(size_t)(16 + l15) * 64];
      v16bf bk0a = ld16(kr0 + khalf,      kr0 + khalf + 16);
      v16bf bk0b = ld16(kr0 + 32 + khalf, kr0 + 48 + khalf);
      v16bf bk1a = ld16(kr1 + khalf,      kr1 + khalf + 16);
      v16bf bk1b = ld16(kr1 + 32 + khalf, kr1 + 48 + khalf);
      s0.v = WMMA_B16(aq0, bk0a, s0.v);
      s0.v = WMMA_B16(aq1, bk0b, s0.v);
      s1.v = WMMA_B16(aq0, bk1a, s1.v);
      s1.v = WMMA_B16(aq1, bk1b, s1.v);
    }

    // ---- causal mask at C-layout -> LDS spill (reference has no 1/sqrt(d))
#pragma unroll
    for (int nt = 0; nt < 2; ++nt) {
      Acc* sp = nt ? &s1 : &s0;
#pragma unroll
      for (int g = 0; g < 8; ++g) {
        int rr = g + 8 * half;
        int cc = nt * 16 + l15;
        float val = ((kt0 + cc) <= (q0 + rr)) ? sp->f[g] : -3.0e38f;
        Ss[w][rr * 32 + cc] = val;
      }
    }
    LDS_FENCE();

    // ---- online softmax: lane pair (l15, l15+16) owns row l15 halves
    {
      int r = l15;
      const float* srow = &Ss[w][r * 32 + half * 16];
      float pm = -3.0e38f;
#pragma unroll
      for (int c = 0; c < 16; ++c) pm = fmaxf(pm, srow[c]);
      float m_it  = fmaxf(pm, __shfl_xor(pm, 16));
      float m_new = fmaxf(mrow, m_it);
      float ps = 0.f;
      bf16* prow = &Ps[w][r * 32 + half * 16];
#pragma unroll
      for (int c = 0; c < 16; ++c) {
        float e = __expf(srow[c] - m_new);
        ps += e;
        prow[c] = (bf16)e;
      }
      float l_it = ps + __shfl_xor(ps, 16);
      float al   = __expf(mrow - m_new);
      lsum = al * lsum + l_it;
      mrow = m_new;
      if (lane < 16) alph[w][r] = al;
    }
    LDS_FENCE();

    // ---- rescale O accumulators by per-row alpha
    float af8[8];
#pragma unroll
    for (int g = 0; g < 8; ++g) af8[g] = alph[w][g + 8 * half];
#pragma unroll
    for (int t = 0; t < 4; ++t)
#pragma unroll
      for (int g = 0; g < 8; ++g) o[t].f[g] *= af8[g];

    // ---- O += P * V  (A from LDS P tile; B from transposed V tile)
    const bf16* pr = &Ps[w][l15 * 32];
    v16bf ap = ld16(pr + khalf, pr + khalf + 16);
#pragma unroll
    for (int dt = 0; dt < 4; ++dt) {
      const bf16* vr = &Vt[p][(dt * 16 + l15) * 32];
      v16bf bv = ld16(vr + khalf, vr + khalf + 16);
      o[dt].v = WMMA_B16(ap, bv, o[dt].v);
    }
  }

  // ---- normalize and write heads-concatenated bf16 output [B,T,H*64]
  if (lane < 16) alph[w][l15] = 1.0f / lsum;
  LDS_FENCE();
  float sc8[8];
#pragma unroll
  for (int g = 0; g < 8; ++g) sc8[g] = alph[w][g + 8 * half];
#pragma unroll
  for (int dt = 0; dt < 4; ++dt)
#pragma unroll
    for (int g = 0; g < 8; ++g) {
      int t = q0 + g + 8 * half;
      int col = h * 64 + dt * 16 + l15;
      outA[((size_t)(b * T + t)) * C + col] = (bf16)(o[dt].f[g] * sc8[g]);
    }
}

// ---------------------------------------------------------------------------
// Host-side orchestration
// ---------------------------------------------------------------------------
extern "C" void kernel_launch(void* const* d_in, const int* in_sizes, int n_in,
                              void* d_out, int out_size, void* d_ws, size_t ws_size,
                              hipStream_t stream) {
  (void)in_sizes; (void)n_in; (void)out_size; (void)ws_size;
  constexpr int Bb = 4, Tt = 2048, Cc = 1024, Hh = 16, Dd = 64, FFf = 4096;
  const int M = Bb * Tt;  // 8192

  const float* x  = (const float*)d_in[0];
  const float* Wq = (const float*)d_in[1];
  const float* Wk = (const float*)d_in[2];
  const float* Wv = (const float*)d_in[3];
  const float* Wo = (const float*)d_in[4];
  const float* bo = (const float*)d_in[5];
  const float* W1 = (const float*)d_in[6];
  const float* b1 = (const float*)d_in[7];
  const float* W2 = (const float*)d_in[8];
  const float* b2 = (const float*)d_in[9];
  const float* g1 = (const float*)d_in[10];
  const float* be1 = (const float*)d_in[11];
  const float* g2 = (const float*)d_in[12];
  const float* be2 = (const float*)d_in[13];
  float* out = (float*)d_out;

  char* p = (char*)d_ws;
  auto alloc = [&](size_t bytes) -> void* {
    void* r = (void*)p;
    p += (bytes + 255) & ~(size_t)255;
    return r;
  };
  const size_t szW   = (size_t)Cc * Cc;  // 1M (also Wq/Wk/Wv reshaped)
  const size_t szAct = (size_t)M * Cc;   // 8M
  const size_t szFF  = (size_t)M * FFf;  // 32M

  bf16* Wq_b  = (bf16*)alloc(szW * 2);
  bf16* Wk_b  = (bf16*)alloc(szW * 2);
  bf16* Wv_b  = (bf16*)alloc(szW * 2);
  bf16* Wo_b  = (bf16*)alloc(szW * 2);
  bf16* W1_b  = (bf16*)alloc((size_t)Cc * FFf * 2);
  bf16* W2_b  = (bf16*)alloc((size_t)FFf * Cc * 2);
  bf16* h1_b  = (bf16*)alloc(szAct * 2);
  bf16* q_b   = (bf16*)alloc(szAct * 2);
  bf16* k_b   = (bf16*)alloc(szAct * 2);
  bf16* v_b   = (bf16*)alloc(szAct * 2);
  bf16* at_b  = (bf16*)alloc(szAct * 2);
  float* x2   = (float*)alloc(szAct * 4);
  bf16* h2_b  = (bf16*)alloc(szAct * 2);
  bf16* ff_b  = (bf16*)alloc(szFF * 2);

  dim3 blk(256);

  // 1) weight conversion (QKV also transposed to [C, H*D])
  wqkv_cvt_kernel<<<dim3(1024), blk, 0, stream>>>(Wq, Wq_b, Cc, Dd, Hh * Dd);
  wqkv_cvt_kernel<<<dim3(1024), blk, 0, stream>>>(Wk, Wk_b, Cc, Dd, Hh * Dd);
  wqkv_cvt_kernel<<<dim3(1024), blk, 0, stream>>>(Wv, Wv_b, Cc, Dd, Hh * Dd);
  f2bf_kernel<<<dim3(1024), blk, 0, stream>>>(Wo, Wo_b, szW);
  f2bf_kernel<<<dim3(1024), blk, 0, stream>>>(W1, W1_b, (size_t)Cc * FFf);
  f2bf_kernel<<<dim3(1024), blk, 0, stream>>>(W2, W2_b, (size_t)FFf * Cc);

  // 2) LN1
  ln_bf16_kernel<<<dim3(M), blk, 0, stream>>>(x, g1, be1, h1_b, Cc);

  // 3) QKV projections -> [B,T,C] bf16 (heads = 64-col slices)
  gemm_wmma_bf16<<<dim3(Cc / 128, M / 128), blk, 0, stream>>>(
      h1_b, Wq_b, nullptr, nullptr, nullptr, q_b, M, Cc, Cc, 0);
  gemm_wmma_bf16<<<dim3(Cc / 128, M / 128), blk, 0, stream>>>(
      h1_b, Wk_b, nullptr, nullptr, nullptr, k_b, M, Cc, Cc, 0);
  gemm_wmma_bf16<<<dim3(Cc / 128, M / 128), blk, 0, stream>>>(
      h1_b, Wv_b, nullptr, nullptr, nullptr, v_b, M, Cc, Cc, 0);

  // 4) causal flash attention -> heads concatenated bf16 [B,T,C]
  attn_flash_kernel<<<dim3(Bb * Hh, Tt / 128), blk, 0, stream>>>(
      q_b, k_b, v_b, at_b, Tt, Cc, Hh);

  // 5) output projection + bias + residual(x) -> x2 fp32
  gemm_wmma_bf16<<<dim3(Cc / 128, M / 128), blk, 0, stream>>>(
      at_b, Wo_b, bo, x, x2, nullptr, M, Cc, Cc, 0);

  // 6) LN2
  ln_bf16_kernel<<<dim3(M), blk, 0, stream>>>(x2, g2, be2, h2_b, Cc);

  // 7) FFN1 + bias + ReLU -> bf16
  gemm_wmma_bf16<<<dim3(FFf / 128, M / 128), blk, 0, stream>>>(
      h2_b, W1_b, b1, nullptr, nullptr, ff_b, M, FFf, Cc, 1);

  // 8) FFN2 + bias + residual(x2) -> d_out fp32
  gemm_wmma_bf16<<<dim3(Cc / 128, M / 128), blk, 0, stream>>>(
      ff_b, W2_b, b2, x2, out, nullptr, M, Cc, FFf, 0);
}
